// HierarchicalDecoder_37873021616457
// MI455X (gfx1250) — matrix operand order repaired and checked by reading, not compiled, not run on hardware
//
#include <hip/hip_runtime.h>
#include <stdint.h>

#define B_DIM 8
#define C_DIM 512
#define S_DIM 2048
#define L_DIM 8921
#define WPB 4                                   // waves (L-tiles) per block
#define LTILES ((L_DIM + 15) / 16)              // 558
#define LT_BLOCKS ((LTILES + WPB - 1) / WPB)    // 140
#define L_PAD (LT_BLOCKS * WPB * 16)            // 8960
#define NIT (S_DIM / 32)                        // 64 S-chunks of 32 rows
#define ROW_HALFS 520                           // padded LDS row stride (1040 B)
#define STAGE_HALFS (32 * ROW_HALFS)            // 16640 halfs = 33280 B
#define STAGE_BYTES (STAGE_HALFS * 2)

typedef __bf16 v16bf __attribute__((ext_vector_type(16)));
typedef __bf16 v8bf  __attribute__((ext_vector_type(8)));
typedef float  v8f   __attribute__((ext_vector_type(8)));

union V16 { v16bf v; v8bf h[2]; };

__device__ __forceinline__ unsigned short f2bf(float f) {
  unsigned int u = __float_as_uint(f);
  u += 0x7FFFu + ((u >> 16) & 1u);       // round-to-nearest-even
  return (unsigned short)(u >> 16);
}

// encoded (B,C,S) f32  ->  Ebf (B,S,C) bf16, LDS-tiled 16x16 transpose
__global__ void transpose_convert(const float* __restrict__ enc,
                                  unsigned short* __restrict__ ebf) {
  __shared__ float tile[16][17];
  const int b  = blockIdx.z;
  const int c0 = blockIdx.y * 16;
  const int s0 = blockIdx.x * 16;
  const int tx = threadIdx.x, ty = threadIdx.y;
  tile[ty][tx] = enc[((size_t)b * C_DIM + (c0 + ty)) * S_DIM + (s0 + tx)];
  __syncthreads();
  ebf[((size_t)b * S_DIM + (s0 + ty)) * C_DIM + (c0 + tx)] = f2bf(tile[tx][ty]);
}

// W_attn / W_cls (L,C) f32 -> (L_PAD,C) bf16, zero-padded tail rows
__global__ void convert_weights(const float* __restrict__ wa,
                                const float* __restrict__ wc,
                                unsigned short* __restrict__ wab,
                                unsigned short* __restrict__ wcb) {
  size_t i = (size_t)blockIdx.x * blockDim.x + threadIdx.x;
  if (i >= (size_t)L_PAD * C_DIM) return;
  if (i < (size_t)L_DIM * C_DIM) {
    wab[i] = f2bf(wa[i]);
    wcb[i] = f2bf(wc[i]);
  } else {
    wab[i] = 0;
    wcb[i] = 0;
  }
}

// Async global->LDS copy of one b128 chunk (GVS mode: SGPR base + VGPR offset)
#define ASYNC_CP(ldsa, voff, sb)                                     \
  asm volatile("global_load_async_to_lds_b128 %0, %1, %2"            \
               :: "v"(ldsa), "v"(voff), "s"(sb) : "memory")

// 4 waves/block on 4 consecutive L-tiles; B (E S-chunks) staged into LDS with
// double-buffered async copies shared by all waves.  Per wave:
// s = E^T Wa^T, t = E^T Wc^T via bf16 WMMA (A register-resident), online
// softmax over S, logit = (sum exp(s)*t)/(sum exp(s)) + b_cls.
__global__ __launch_bounds__(128) void decoder_main(
    const unsigned short* __restrict__ Ebf,
    const unsigned short* __restrict__ Wa,
    const unsigned short* __restrict__ Wc,
    const float* __restrict__ bcls,
    float* __restrict__ out) {
  __shared__ unsigned short ldsB[2 * STAGE_HALFS];   // 66,560 B

  const int tid   = threadIdx.x;
  const int lane  = tid & 31;
  const int wv    = tid >> 5;
  const int ltile = blockIdx.x * WPB + wv;
  const int b     = blockIdx.y;
  const int lbase = ltile * 16;

  const int l0 = lane & 15;   // A row (M) and B column (N) for this lane
  const int kh = lane >> 4;   // K sub-chunk select per ISA VGPR layout

  // Loop-invariant A fragments: 2 matrices x 16 K-steps -> registers
  V16 Aa[16], Ac[16];
  {
    const unsigned short* pA = Wa + (size_t)(lbase + l0) * C_DIM + kh * 8;
    const unsigned short* pC = Wc + (size_t)(lbase + l0) * C_DIM + kh * 8;
#pragma unroll
    for (int k = 0; k < 16; ++k) {
      Aa[k].h[0] = *(const v8bf*)(pA + k * 32);
      Aa[k].h[1] = *(const v8bf*)(pA + k * 32 + 16);
      Ac[k].h[0] = *(const v8bf*)(pC + k * 32);
      Ac[k].h[1] = *(const v8bf*)(pC + k * 32 + 16);
    }
  }

  // Async-copy parameters (pointer escapes via the int cast, so the compiler
  // must honor the asm memory clobber for LDS reads).
  const unsigned ldsBase =
      (unsigned)(uintptr_t)(void*)ldsB;  // raw LDS byte offset (addr[31:0])
  const unsigned long long gBase =
      (unsigned long long)(uintptr_t)(Ebf + (size_t)b * S_DIM * C_DIM);
  const unsigned c16  = (unsigned)(tid & 63) * 16u;  // 16B chunk within 2 rows
  const int      rsel = tid >> 6;                    // which of the 2 rows

  auto issue_stage = [&](int it) {
    const unsigned stageOff = ldsBase + (unsigned)(it & 1) * (unsigned)STAGE_BYTES;
    const unsigned gIt      = (unsigned)it * 32768u;  // 32 rows * 1024 B
#pragma unroll
    for (int j = 0; j < 16; ++j) {                    // 2 rows per round
      const unsigned r    = (unsigned)(2 * j + rsel);
      const unsigned ldsa = stageOff + r * 1040u + c16;
      const unsigned voff = gIt + r * 1024u + c16;
      ASYNC_CP(ldsa, voff, gBase);
    }
  };

  float m[8], z[8], acc[8];
#pragma unroll
  for (int r = 0; r < 8; ++r) { m[r] = -3.0e38f; z[r] = 0.f; acc[r] = 0.f; }

  issue_stage(0);
  for (int it = 0; it < NIT; ++it) {
    if (it + 1 < NIT) {
      issue_stage(it + 1);
      asm volatile("s_wait_asynccnt 16" ::: "memory");  // stage `it` landed
    } else {
      asm volatile("s_wait_asynccnt 0" ::: "memory");
    }
    __syncthreads();  // publish stage `it` to all waves

    const unsigned short* sb0 =
        ldsB + (size_t)(it & 1) * STAGE_HALFS + (size_t)l0 * ROW_HALFS + kh * 8;
    const unsigned short* sb1 = sb0 + 16 * ROW_HALFS;

    v8f c0s = {}; v8f c0t = {}; v8f c1s = {}; v8f c1t = {};
#pragma unroll
    for (int k = 0; k < 16; ++k) {
      V16 b0, b1;
      b0.h[0] = *(const v8bf*)(sb0 + k * 32);
      b0.h[1] = *(const v8bf*)(sb0 + k * 32 + 16);
      b1.h[0] = *(const v8bf*)(sb1 + k * 32);
      b1.h[1] = *(const v8bf*)(sb1 + k * 32 + 16);
      c0s = __builtin_amdgcn_wmma_f32_16x16x32_bf16(
          false, Aa[k].v, false, b0.v, (short)0, c0s, false, false);
      c0t = __builtin_amdgcn_wmma_f32_16x16x32_bf16(
          false, Ac[k].v, false, b0.v, (short)0, c0t, false, false);
      c1s = __builtin_amdgcn_wmma_f32_16x16x32_bf16(
          false, Aa[k].v, false, b1.v, (short)0, c1s, false, false);
      c1t = __builtin_amdgcn_wmma_f32_16x16x32_bf16(
          false, Ac[k].v, false, b1.v, (short)0, c1t, false, false);
    }

    // Online softmax update (16 S-columns per row spread across lanes)
#pragma unroll
    for (int r = 0; r < 8; ++r) {
      const float s  = c0s[r];
      const float mo = m[r];
      const float mn = fmaxf(mo, s);
      const float e  = __expf(s - mn);
      const float sc = __expf(mo - mn);
      z[r]   = z[r] * sc + e;
      acc[r] = acc[r] * sc + e * c0t[r];
      m[r]   = mn;
    }
#pragma unroll
    for (int r = 0; r < 8; ++r) {
      const float s  = c1s[r];
      const float mo = m[r];
      const float mn = fmaxf(mo, s);
      const float e  = __expf(s - mn);
      const float sc = __expf(mo - mn);
      z[r]   = z[r] * sc + e;
      acc[r] = acc[r] * sc + e * c1t[r];
      m[r]   = mn;
    }

    __syncthreads();  // all waves done reading before this buffer is refilled
  }

  // Cross-lane merge of (m, z, acc) within each 16-lane group (per row)
  float myres = 0.f;
#pragma unroll
  for (int r = 0; r < 8; ++r) {
    float mv = m[r], zv = z[r], av = acc[r];
#pragma unroll
    for (int mask = 1; mask <= 8; mask <<= 1) {
      const float m2 = __shfl_xor(mv, mask, 32);
      const float z2 = __shfl_xor(zv, mask, 32);
      const float a2 = __shfl_xor(av, mask, 32);
      const float mn = fmaxf(mv, m2);
      const float s1 = __expf(mv - mn);
      const float s2 = __expf(m2 - mn);
      zv = zv * s1 + z2 * s2;
      av = av * s1 + a2 * s2;
      mv = mn;
    }
    if ((lane & 7) == r) myres = av / zv;
  }
  // Accumulator VGPR r holds rows M=r (lanes 0-15) and M=r+8 (lanes 16-31)
  if ((lane & 15) < 8) {
    const int row = lbase + (lane & 7) + 8 * kh;
    if (row < L_DIM) out[(size_t)b * L_DIM + row] = myres + bcls[row];
  }
}

extern "C" void kernel_launch(void* const* d_in, const int* in_sizes, int n_in,
                              void* d_out, int out_size, void* d_ws,
                              size_t ws_size, hipStream_t stream) {
  (void)in_sizes; (void)n_in; (void)out_size; (void)ws_size;
  const float* enc = (const float*)d_in[0];
  const float* Wa  = (const float*)d_in[1];
  // d_in[2] = b_attn: softmax over S is shift-invariant per (b,l) -> unused
  const float* Wc  = (const float*)d_in[3];
  const float* bc  = (const float*)d_in[4];
  float* out = (float*)d_out;

  unsigned char* ws = (unsigned char*)d_ws;
  const size_t szE = (size_t)B_DIM * S_DIM * C_DIM * sizeof(unsigned short);
  const size_t szW = (size_t)L_PAD * C_DIM * sizeof(unsigned short);
  const size_t oWa = (szE + 255) & ~(size_t)255;
  const size_t oWc = oWa + ((szW + 255) & ~(size_t)255);
  unsigned short* Ebf = (unsigned short*)ws;
  unsigned short* Wab = (unsigned short*)(ws + oWa);
  unsigned short* Wcb = (unsigned short*)(ws + oWc);

  hipLaunchKernelGGL(transpose_convert, dim3(S_DIM / 16, C_DIM / 16, B_DIM),
                     dim3(16, 16), 0, stream, enc, Ebf);
  const size_t nW = (size_t)L_PAD * C_DIM;
  hipLaunchKernelGGL(convert_weights, dim3((unsigned)((nW + 255) / 256)),
                     dim3(256), 0, stream, Wa, Wc, Wab, Wcb);
  hipLaunchKernelGGL(decoder_main, dim3(LT_BLOCKS, B_DIM), dim3(128), 0,
                     stream, Ebf, Wab, Wcb, bc, out);
}
